// TreeGRU_66176856097329
// MI455X (gfx1250) — compile-verified, matrix operand order checked by reading, not run on hardware
//
#include <hip/hip_runtime.h>
#include <hip/hip_bf16.h>
#include <math.h>

typedef __attribute__((ext_vector_type(16))) _Float16 v16h;
typedef __attribute__((ext_vector_type(8)))  _Float16 v8h;
typedef __attribute__((ext_vector_type(8)))  float    v8f;

#define H    256
#define LDX  264          // padded LDS row stride (halfs): 528B -> bank-spread
#define BM   32           // parent rows per block
#define NW   32           // 32 waves: (2 m-subtiles) x (16 c-slices)
#define TPB  (NW * 32)    // 1024 threads

#if __has_builtin(__builtin_amdgcn_tanhf)
#define TANHF(x) __builtin_amdgcn_tanhf(x)
#else
#define TANHF(x) tanhf(x)
#endif
#if __has_builtin(__builtin_amdgcn_rcpf)
#define RCPF(x) __builtin_amdgcn_rcpf(x)
#else
#define RCPF(x) (1.0f / (x))
#endif

__device__ __forceinline__ float sigmoidf_(float x) {
    return RCPF(1.0f + __expf(-x));
}

// Build a 16x16x32 f16 WMMA operand fragment: lane (L<16) holds K in
// [k0, k0+8) and [k0+16, k0+24); lane (L>=16) holds the +8-shifted halves.
// Two 16B loads (ds_load_b128 / global_load_b128).
__device__ __forceinline__ v16h load_frag(const _Float16* __restrict__ row,
                                          int k0, int hi) {
    v16h f;
    *((v8h*)&f)       = *(const v8h*)(row + k0 + hi * 8);
    *(((v8h*)&f) + 1) = *(const v8h*)(row + k0 + 16 + hi * 8);
    return f;
}

__device__ __forceinline__ v8f wmma_f16(v16h a, v16h b, v8f c) {
    return __builtin_amdgcn_wmma_f32_16x16x32_f16(
        /*neg_a=*/false, a, /*neg_b=*/false, b,
        /*c_mod=*/(short)0, c, /*reuse_a=*/false, /*reuse_b=*/false);
}

// ---- Weight conversion: fp32 [768,256] row-major -> f16, same layout ----
__global__ void treegru_wcvt(const float* __restrict__ wih,
                             const float* __restrict__ whh,
                             _Float16* __restrict__ w16) {
    int i = blockIdx.x * blockDim.x + threadIdx.x;   // 0 .. 2*196608-1
    const int WN = 3 * H * H;                        // 196608
    if (i < WN)            w16[i] = (_Float16)wih[i];
    else if (i < 2 * WN)   w16[i] = (_Float16)whh[i - WN];
}

// ---- One tree level: parents[p] = GRU(GRU(0, child0), child1) ----
__global__ __launch_bounds__(TPB)
void treegru_level(const _Float16* __restrict__ hin,   // null on level0
                   const int*      __restrict__ tok,   // level0 only
                   const float*    __restrict__ emb,   // level0 only
                   const _Float16* __restrict__ w16,   // [wih f16 | whh f16]
                   const float*    __restrict__ b_ih,
                   const float*    __restrict__ b_hh,
                   _Float16*       __restrict__ hout16,
                   float*          __restrict__ hout32,
                   int N, int level0, int outf32) {
    __shared__ _Float16 lds_x[2 * BM * LDX];   // 64 child rows (f16)
    __shared__ _Float16 lds_h1[BM * LDX];      // 32 intermediate rows

    const int tid   = threadIdx.x;
    const int lane  = tid & 31;
    const int wv    = tid >> 5;          // 0..31
    const int mt    = wv >> 4;           // m-subtile 0..1
    const int cw    = wv & 15;           // c-slice 0..15
    const int halfN = N >> 1;
    const int pr0   = blockIdx.x * BM;   // first parent row this block

    // ---------------- Phase A: stage 2*BM child rows into LDS (f16) -------
    if (level0) {
        for (int idx = tid; idx < 2 * BM * (H / 4); idx += TPB) {
            int r  = idx >> 6;              // child slot 0..63
            int c4 = (idx & 63) << 2;       // 4-float column group
            int pr = pr0 + (r >> 1);
            int b  = pr / halfN;
            int p  = pr - b * halfN;
            int n  = (p << 1) + (r & 1);
            int t  = tok[(size_t)b * N + n];
            float4 v = *(const float4*)(emb + (size_t)t * H + c4);
            _Float16* d = lds_x + r * LDX + c4;
            d[0] = (_Float16)TANHF(v.x);
            d[1] = (_Float16)TANHF(v.y);
            d[2] = (_Float16)TANHF(v.z);
            d[3] = (_Float16)TANHF(v.w);
        }
    } else {
        for (int idx = tid; idx < 2 * BM * (H / 8); idx += TPB) {
            int r  = idx >> 5;
            int c8 = (idx & 31) << 3;       // 8-half (16B) column group
            int pr = pr0 + (r >> 1);
            int b  = pr / halfN;
            int p  = pr - b * halfN;
            size_t crow = (size_t)b * N + (p << 1) + (r & 1);
            *(v8h*)(lds_x + r * LDX + c8) =
                *(const v8h*)(hin + crow * H + c8);
        }
    }
    __syncthreads();

    // ---------------- Per-wave tile setup ---------------------------------
    const int nl  = lane & 15;
    const int hi  = lane >> 4;
    const int c0  = cw * 16;
    const int col = c0 + nl;             // gate column owned by this lane

    const _Float16* wih = w16;
    const _Float16* whh = w16 + 3 * H * H;
    const _Float16* wr = wih + (size_t)(0 * H + col) * H;  // W rows = B cols
    const _Float16* wz = wih + (size_t)(1 * H + col) * H;
    const _Float16* wn = wih + (size_t)(2 * H + col) * H;
    const _Float16* ur = whh + (size_t)(0 * H + col) * H;
    const _Float16* uz = whh + (size_t)(1 * H + col) * H;
    const _Float16* un = whh + (size_t)(2 * H + col) * H;

    const float bir = b_ih[col], biz = b_ih[H + col], bin_ = b_ih[2 * H + col];
    const float bhr = b_hh[col], bhz = b_hh[H + col], bhn  = b_hh[2 * H + col];

    // ---------------- Phase B: child0 step (h_prev = 0 => gh = b_hh) ------
    {
        const _Float16* arow = lds_x + (2 * (mt * 16 + nl)) * LDX;  // child0
        v8f ar = {}, az = {}, an = {};
#pragma unroll
        for (int kk = 0; kk < H / 32; ++kk) {
            int k0 = kk * 32;
            v16h a  = load_frag(arow, k0, hi);
            v16h br = load_frag(wr, k0, hi);
            v16h bz = load_frag(wz, k0, hi);
            v16h bn = load_frag(wn, k0, hi);
            ar = wmma_f16(a, br, ar);
            az = wmma_f16(a, bz, az);
            an = wmma_f16(a, bn, an);
        }
#pragma unroll
        for (int j = 0; j < 8; ++j) {
            float r  = sigmoidf_(ar[j] + bir + bhr);
            float z  = sigmoidf_(az[j] + biz + bhz);
            float n  = TANHF(an[j] + bin_ + r * bhn);
            float h1 = (1.0f - z) * n;          // + z*0
            lds_h1[(mt * 16 + j + 8 * hi) * LDX + col] = (_Float16)h1;
        }
    }
    __syncthreads();

    // ---------------- Phase C: child1 step (h_prev = h1) ------------------
    {
        const _Float16* axrow = lds_x + (2 * (mt * 16 + nl) + 1) * LDX; // child1
        const _Float16* ahrow = lds_h1 + (mt * 16 + nl) * LDX;
        v8f ir = {}, iz = {}, in_ = {}, hr = {}, hz = {}, hn = {};
#pragma unroll
        for (int kk = 0; kk < H / 32; ++kk) {
            int k0 = kk * 32;
            v16h ax = load_frag(axrow, k0, hi);
            v16h ah = load_frag(ahrow, k0, hi);
            v16h br = load_frag(wr, k0, hi);
            v16h bz = load_frag(wz, k0, hi);
            v16h bn = load_frag(wn, k0, hi);
            v16h cr = load_frag(ur, k0, hi);
            v16h cz = load_frag(uz, k0, hi);
            v16h cn = load_frag(un, k0, hi);
            ir  = wmma_f16(ax, br, ir);
            iz  = wmma_f16(ax, bz, iz);
            in_ = wmma_f16(ax, bn, in_);
            hr  = wmma_f16(ah, cr, hr);
            hz  = wmma_f16(ah, cz, hz);
            hn  = wmma_f16(ah, cn, hn);
        }
#pragma unroll
        for (int j = 0; j < 8; ++j) {
            int   m  = mt * 16 + j + 8 * hi;
            float h1 = (float)lds_h1[m * LDX + col];
            float r  = sigmoidf_(ir[j] + hr[j] + bir + bhr);
            float z  = sigmoidf_(iz[j] + hz[j] + biz + bhz);
            float n  = TANHF(in_[j] + bin_ + r * (hn[j] + bhn));
            float h2 = (1.0f - z) * n + z * h1;
            size_t prow = (size_t)pr0 + m;
            if (outf32) hout32[prow * H + col] = h2;
            else        hout16[prow * H + col] = (_Float16)h2;
        }
    }
}

extern "C" void kernel_launch(void* const* d_in, const int* in_sizes, int n_in,
                              void* d_out, int out_size, void* d_ws, size_t ws_size,
                              hipStream_t stream) {
    const int*   tok = (const int*)d_in[0];
    const float* emb = (const float*)d_in[1];
    const float* wih = (const float*)d_in[2];
    const float* whh = (const float*)d_in[3];
    const float* bih = (const float*)d_in[4];
    const float* bhh = (const float*)d_in[5];
    float* out = (float*)d_out;

    char* ws = (char*)d_ws;
    _Float16* w16 = (_Float16*)ws;                                // 768 KB
    _Float16* hA  = (_Float16*)(ws + (size_t)(1u << 20));         // <= 67.2 MB
    _Float16* hB  = (_Float16*)(ws + (size_t)(1u << 20) + (size_t)(70u << 20));

    // fp32 -> f16 weight conversion (2 * 768 * 256 elements)
    hipLaunchKernelGGL(treegru_wcvt, dim3((2 * 3 * H * H) / 512), dim3(512),
                       0, stream, wih, whh, w16);

    const int B = 64;
    int N = 4096;
    const _Float16* hin = nullptr;
    _Float16* bufs[2] = {hA, hB};
    int cur = 0, level0 = 1;

    while (N > 1) {
        int parents = B * (N >> 1);
        int outf32  = (N == 2);
        _Float16* ho = bufs[cur];
        hipLaunchKernelGGL(treegru_level, dim3(parents / BM), dim3(TPB), 0, stream,
                           hin, tok, emb, w16, bih, bhh,
                           outf32 ? (_Float16*)nullptr : ho,
                           outf32 ? out : (float*)nullptr,
                           N, level0, outf32);
        hin = ho;
        cur ^= 1;
        level0 = 0;
        N >>= 1;
    }
}
